// Transformer_16458314678688
// MI455X (gfx1250) — matrix operand order repaired and checked by reading, not compiled, not run on hardware
//
#include <hip/hip_runtime.h>

// ---------------------------------------------------------------------------
// Types for CDNA5 WMMA (wave32): v_wmma_f32_16x16x32_bf16
// ---------------------------------------------------------------------------
typedef __attribute__((ext_vector_type(16))) __bf16       v16bf;
typedef __attribute__((ext_vector_type(8)))  float        v8f;
typedef __attribute__((ext_vector_type(4)))  unsigned int v4u;

union Frag { v4u u[2]; v16bf v; };

__device__ __forceinline__ unsigned short f2bf(float f) {
  union { float f; unsigned int u; } c; c.f = f;
  unsigned int r = c.u + 0x7FFFu + ((c.u >> 16) & 1u);
  return (unsigned short)(r >> 16);
}

__device__ __forceinline__ v8f wmma_bf16(Frag a, Frag b, v8f acc) {
  return __builtin_amdgcn_wmma_f32_16x16x32_bf16(
      false, a.v, false, b.v, (short)0, acc, false, false);
}

// ---------------------------------------------------------------------------
// Weight convert+transpose: fp32 W[K x N] -> bf16 Wt[N x K]
// ---------------------------------------------------------------------------
__global__ void k_w_bf16t(const float* __restrict__ W,
                          unsigned short* __restrict__ Wt, int K, int N) {
  int idx = blockIdx.x * 256 + threadIdx.x;
  if (idx >= K * N) return;
  int n = idx % N, k = idx / N;
  Wt[(size_t)n * K + k] = f2bf(W[idx]);
}

// V transpose: bf16 V[B*L, 512] -> bf16 Vt[(b*8+h)*64+d][L]
__global__ void k_v_t(const unsigned short* __restrict__ V,
                      unsigned short* __restrict__ Vt, int L) {
  int idx = blockIdx.x * 256 + threadIdx.x;   // B*L*512 total
  int c = idx & 511, row = idx >> 9;
  int l = row % L, b = row / L;
  int h = c >> 6, d = c & 63;
  Vt[(size_t)((b * 8 + h) * 64 + d) * L + l] = V[idx];
}

// ---------------------------------------------------------------------------
// Input projection: x = pos_emb + inp @ W + bias  (K tiny: 16 or 8)
// ---------------------------------------------------------------------------
__global__ void k_input_proj(const float* __restrict__ inp,
                             const float* __restrict__ pos,
                             const float* __restrict__ W,
                             const float* __restrict__ bias,
                             float* __restrict__ xf,
                             unsigned short* __restrict__ xb, int Kin) {
  int idx = blockIdx.x * 256 + threadIdx.x;   // 4096*512
  int col = idx & 511, row = idx >> 9, l = row & 511;
  float acc = pos[(size_t)l * 512 + col] + bias[col];
  for (int k = 0; k < Kin; ++k)
    acc += inp[(size_t)row * Kin + k] * W[(size_t)k * 512 + col];
  xf[idx] = acc;
  xb[idx] = f2bf(acc);
}

// ---------------------------------------------------------------------------
// WMMA GEMM: out[M x N] = A_bf16[M x K] * Bt_bf16[N x K]^T (+bias)(+relu)(+resid)
// 8 waves/WG; each wave: 16(M) x 64(N) tile, K-step 32, double-buffered
// fragments so loads for chunk k+32 overlap WMMAs of chunk k.
// Requires K % 64 == 0 (K is 512 or 2048 here).
// ---------------------------------------------------------------------------
__global__ void __launch_bounds__(256)
k_gemm_wmma(const unsigned short* __restrict__ A,
            const unsigned short* __restrict__ Bt,
            const float* __restrict__ bias,
            const float* __restrict__ resid,
            float* __restrict__ outf,
            unsigned short* __restrict__ outb,
            int N, int K, int relu) {
  const int lane = threadIdx.x & 31;
  const int wave = threadIdx.x >> 5;
  const int l16  = lane & 15;
  const int half = lane >> 4;
  const int m0   = blockIdx.y * 128 + wave * 16;
  const int n0   = blockIdx.x * 64;

  v8f acc[4];
#pragma unroll
  for (int c = 0; c < 4; ++c)
#pragma unroll
    for (int i = 0; i < 8; ++i) acc[c][i] = 0.0f;

  const unsigned short* arow = A  + (size_t)(m0 + l16) * K + (half ? 8 : 0);
  const unsigned short* brow = Bt + (size_t)(n0 + l16) * K + (half ? 8 : 0);

  Frag aA, aB, bA[4], bB[4];

  // prologue: chunk 0 -> buffer A
  aA.u[0] = *(const v4u*)(arow);
  aA.u[1] = *(const v4u*)(arow + 16);
#pragma unroll
  for (int c = 0; c < 4; ++c) {
    const unsigned short* p = brow + (size_t)(c * 16) * K;
    bA[c].u[0] = *(const v4u*)(p);
    bA[c].u[1] = *(const v4u*)(p + 16);
  }

  int k = 0;
  for (; k + 64 < K; k += 64) {
    // prefetch chunk k+32 -> buffer B
    aB.u[0] = *(const v4u*)(arow + k + 32);
    aB.u[1] = *(const v4u*)(arow + k + 48);
#pragma unroll
    for (int c = 0; c < 4; ++c) {
      const unsigned short* p = brow + (size_t)(c * 16) * K + k + 32;
      bB[c].u[0] = *(const v4u*)(p);
      bB[c].u[1] = *(const v4u*)(p + 16);
    }
#pragma unroll
    for (int c = 0; c < 4; ++c) acc[c] = wmma_bf16(aA, bA[c], acc[c]);

    // prefetch chunk k+64 -> buffer A
    aA.u[0] = *(const v4u*)(arow + k + 64);
    aA.u[1] = *(const v4u*)(arow + k + 80);
#pragma unroll
    for (int c = 0; c < 4; ++c) {
      const unsigned short* p = brow + (size_t)(c * 16) * K + k + 64;
      bA[c].u[0] = *(const v4u*)(p);
      bA[c].u[1] = *(const v4u*)(p + 16);
    }
#pragma unroll
    for (int c = 0; c < 4; ++c) acc[c] = wmma_bf16(aB, bB[c], acc[c]);
  }

  // tail: k == K-64
  aB.u[0] = *(const v4u*)(arow + k + 32);
  aB.u[1] = *(const v4u*)(arow + k + 48);
#pragma unroll
  for (int c = 0; c < 4; ++c) {
    const unsigned short* p = brow + (size_t)(c * 16) * K + k + 32;
    bB[c].u[0] = *(const v4u*)(p);
    bB[c].u[1] = *(const v4u*)(p + 16);
  }
#pragma unroll
  for (int c = 0; c < 4; ++c) acc[c] = wmma_bf16(aA, bA[c], acc[c]);
#pragma unroll
  for (int c = 0; c < 4; ++c) acc[c] = wmma_bf16(aB, bB[c], acc[c]);

#pragma unroll
  for (int c = 0; c < 4; ++c) {
    const int col = n0 + c * 16 + l16;
    const float bv = bias ? bias[col] : 0.0f;
#pragma unroll
    for (int i = 0; i < 8; ++i) {
      const int row = m0 + i + half * 8;
      float v = acc[c][i] + bv;
      if (relu) v = fmaxf(v, 0.0f);
      size_t o = (size_t)row * N + col;
      if (resid) v += resid[o];
      if (outf) outf[o] = v;
      if (outb) outb[o] = f2bf(v);
    }
  }
}

// ---------------------------------------------------------------------------
// Flash attention (online softmax). One wave = one (b,h,16-query) tile.
// Q,K bf16 [B*L,512] row-major; Vt bf16 [(b*8+h)*64+d][L].
// kvlen < L implements the decoder column mask (last PRED_LEN cols = -inf).
// All 8 global fragments (4 K + 4 V) are issued up front each iteration so
// the V loads stay in flight across the score WMMAs + softmax VALU work.
// ---------------------------------------------------------------------------
__global__ void __launch_bounds__(256)
k_attn(const unsigned short* __restrict__ Qm,
       const unsigned short* __restrict__ Km,
       const unsigned short* __restrict__ Vt,
       float* __restrict__ ctxf,
       unsigned short* __restrict__ ctxb,
       int L, int kvlen) {
  __shared__ unsigned short pbuf[8][16 * 32];

  const int lane = threadIdx.x & 31;
  const int wave = threadIdx.x >> 5;
  const int l16  = lane & 15;
  const int half = lane >> 4;
  const int bh = blockIdx.x;          // b*8 + h
  const int b = bh >> 3, h = bh & 7;
  const int q0 = blockIdx.y * 128 + wave * 16;

  // Q A-fragments: d 0..31 and 32..63
  Frag a0, a1;
  {
    const unsigned short* qr =
        Qm + (size_t)(b * L + q0 + l16) * 512 + h * 64 + (half ? 8 : 0);
    a0.u[0] = *(const v4u*)(qr);
    a0.u[1] = *(const v4u*)(qr + 16);
    a1.u[0] = *(const v4u*)(qr + 32);
    a1.u[1] = *(const v4u*)(qr + 48);
  }

  float mrow[8], lrow[8];
  v8f o[4];
#pragma unroll
  for (int i = 0; i < 8; ++i) { mrow[i] = -1e30f; lrow[i] = 0.0f; }
#pragma unroll
  for (int c = 0; c < 4; ++c)
#pragma unroll
    for (int i = 0; i < 8; ++i) o[c][i] = 0.0f;

  for (int k0 = 0; k0 < kvlen; k0 += 32) {
    // ---- issue all global fragment loads for this key block ---------------
    Frag bk[4], bv[4];
    {
      const unsigned short* kr =
          Km + (size_t)(b * L + k0 + l16) * 512 + h * 64 + (half ? 8 : 0);
      const unsigned short* kr1 = kr + (size_t)16 * 512;
      bk[0].u[0] = *(const v4u*)(kr);       bk[0].u[1] = *(const v4u*)(kr + 16);
      bk[1].u[0] = *(const v4u*)(kr + 32);  bk[1].u[1] = *(const v4u*)(kr + 48);
      bk[2].u[0] = *(const v4u*)(kr1);      bk[2].u[1] = *(const v4u*)(kr1 + 16);
      bk[3].u[0] = *(const v4u*)(kr1 + 32); bk[3].u[1] = *(const v4u*)(kr1 + 48);
    }
#pragma unroll
    for (int c = 0; c < 4; ++c) {
      const unsigned short* vr =
          Vt + (size_t)(bh * 64 + c * 16 + l16) * L + k0 + (half ? 8 : 0);
      bv[c].u[0] = *(const v4u*)(vr);
      bv[c].u[1] = *(const v4u*)(vr + 16);
    }

    // ---- scores: S(16q x 32keys) = Q(16x64) x K^T -------------------------
    v8f s0, s1;
#pragma unroll
    for (int i = 0; i < 8; ++i) { s0[i] = 0.0f; s1[i] = 0.0f; }
    s0 = wmma_bf16(a0, bk[0], s0);
    s0 = wmma_bf16(a1, bk[1], s0);
    s1 = wmma_bf16(a0, bk[2], s1);
    s1 = wmma_bf16(a1, bk[3], s1);

    // ---- online softmax update (per-row stats in 16-lane groups) ----------
    float alpha[8];
#pragma unroll
    for (int i = 0; i < 8; ++i) {
      float x0 = s0[i] * 0.125f;        // 1/sqrt(64)
      float x1 = s1[i] * 0.125f;
      float t = fmaxf(x0, x1);
      t = fmaxf(t, __shfl_xor(t, 1, 32));
      t = fmaxf(t, __shfl_xor(t, 2, 32));
      t = fmaxf(t, __shfl_xor(t, 4, 32));
      t = fmaxf(t, __shfl_xor(t, 8, 32));
      float mn = fmaxf(mrow[i], t);
      float p0 = __expf(x0 - mn);
      float p1 = __expf(x1 - mn);
      float r = p0 + p1;
      r += __shfl_xor(r, 1, 32);
      r += __shfl_xor(r, 2, 32);
      r += __shfl_xor(r, 4, 32);
      r += __shfl_xor(r, 8, 32);
      alpha[i] = __expf(mrow[i] - mn);
      lrow[i] = lrow[i] * alpha[i] + r;
      mrow[i] = mn;
      int row = i + half * 8;
      pbuf[wave][row * 32 + l16]      = f2bf(p0);
      pbuf[wave][row * 32 + 16 + l16] = f2bf(p1);
    }

#pragma unroll
    for (int c = 0; c < 4; ++c)
#pragma unroll
      for (int i = 0; i < 8; ++i) o[c][i] *= alpha[i];

    // ---- P (D-layout) -> A-fragment via per-wave LDS tile -----------------
    Frag ap;
    {
      const unsigned short* pp = &pbuf[wave][l16 * 32 + (half ? 8 : 0)];
      ap.u[0] = *(const v4u*)(pp);
      ap.u[1] = *(const v4u*)(pp + 16);
    }

    // ---- ctx += P(16x32) x V(32x64) ---------------------------------------
#pragma unroll
    for (int c = 0; c < 4; ++c) o[c] = wmma_bf16(ap, bv[c], o[c]);
  }

  // ---- finalize: divide by row sums, write fp32 + bf16 --------------------
#pragma unroll
  for (int c = 0; c < 4; ++c) {
    const int col = h * 64 + c * 16 + l16;
#pragma unroll
    for (int i = 0; i < 8; ++i) {
      const int row = q0 + i + half * 8;
      float v = o[c][i] / lrow[i];
      size_t off = (size_t)(b * L + row) * 512 + col;
      ctxf[off] = v;
      ctxb[off] = f2bf(v);
    }
  }
}

// ---------------------------------------------------------------------------
// LayerNorm over rows of 512; one wave per row; writes fp32 + bf16.
// ---------------------------------------------------------------------------
__global__ void __launch_bounds__(256)
k_layernorm(const float* __restrict__ x, const float* __restrict__ g,
            const float* __restrict__ bb, float* __restrict__ yf,
            unsigned short* __restrict__ yb) {
  const int row  = blockIdx.x * 8 + (threadIdx.x >> 5);
  const int lane = threadIdx.x & 31;
  const float* xr = x + (size_t)row * 512;
  float v[16];
  float s = 0.0f;
#pragma unroll
  for (int i = 0; i < 16; ++i) { v[i] = xr[lane + i * 32]; s += v[i]; }
  s += __shfl_xor(s, 1, 32);  s += __shfl_xor(s, 2, 32);
  s += __shfl_xor(s, 4, 32);  s += __shfl_xor(s, 8, 32);
  s += __shfl_xor(s, 16, 32);
  const float mu = s * (1.0f / 512.0f);
  float vs = 0.0f;
#pragma unroll
  for (int i = 0; i < 16; ++i) { float d = v[i] - mu; vs += d * d; }
  vs += __shfl_xor(vs, 1, 32);  vs += __shfl_xor(vs, 2, 32);
  vs += __shfl_xor(vs, 4, 32);  vs += __shfl_xor(vs, 8, 32);
  vs += __shfl_xor(vs, 16, 32);
  const float rstd = rsqrtf(vs * (1.0f / 512.0f) + 1e-5f);
#pragma unroll
  for (int i = 0; i < 16; ++i) {
    int col = lane + i * 32;
    float o = (v[i] - mu) * rstd * g[col] + bb[col];
    size_t off = (size_t)row * 512 + col;
    yf[off] = o;
    yb[off] = f2bf(o);
  }
}

// ---------------------------------------------------------------------------
// Final projection: out[row, 0..15] = x[row]·proj_w + proj_b (fp32)
// ---------------------------------------------------------------------------
__global__ void k_final_proj(const float* __restrict__ x,
                             const float* __restrict__ W,
                             const float* __restrict__ bias,
                             float* __restrict__ out) {
  int idx = blockIdx.x * 256 + threadIdx.x;   // 4096*16
  if (idx >= 4096 * 16) return;
  int j = idx & 15, row = idx >> 4;
  float acc = bias[j];
  for (int k = 0; k < 512; ++k)
    acc += x[(size_t)row * 512 + k] * W[k * 16 + j];
  out[idx] = acc;
}

// ---------------------------------------------------------------------------
// Host orchestration
// ---------------------------------------------------------------------------
extern "C" void kernel_launch(void* const* d_in, const int* in_sizes, int n_in,
                              void* d_out, int out_size, void* d_ws,
                              size_t ws_size, hipStream_t stream) {
  (void)in_sizes; (void)n_in; (void)out_size; (void)ws_size;
  const int Bz = 8, L = 512, D = 512, F = 2048, H = 8;
  const int M = Bz * L;   // 4096

  // inputs
  const float* src         = (const float*)d_in[0];
  const float* tgt         = (const float*)d_in[1];
  const float* src_pos     = (const float*)d_in[2];
  const float* tgt_pos     = (const float*)d_in[3];
  const float* src_lin_w   = (const float*)d_in[4];
  const float* src_lin_b   = (const float*)d_in[5];
  const float* tgt_lin_w   = (const float*)d_in[6];
  const float* tgt_lin_b   = (const float*)d_in[7];
  const float* enc_wq      = (const float*)d_in[8];
  const float* enc_wk      = (const float*)d_in[9];
  const float* enc_wv      = (const float*)d_in[10];
  const float* enc_wo      = (const float*)d_in[11];
  const float* enc_ff1_w   = (const float*)d_in[12];
  const float* enc_ff1_b   = (const float*)d_in[13];
  const float* enc_ff2_w   = (const float*)d_in[14];
  const float* enc_ff2_b   = (const float*)d_in[15];
  const float* enc_ln1_g   = (const float*)d_in[16];
  const float* enc_ln1_b   = (const float*)d_in[17];
  const float* enc_ln2_g   = (const float*)d_in[18];
  const float* enc_ln2_b   = (const float*)d_in[19];
  const float* dec_sa_wq   = (const float*)d_in[20];
  const float* dec_sa_wk   = (const float*)d_in[21];
  const float* dec_sa_wv   = (const float*)d_in[22];
  const float* dec_sa_wo   = (const float*)d_in[23];
  const float* dec_ca_wq   = (const float*)d_in[24];
  const float* dec_ca_wk   = (const float*)d_in[25];
  const float* dec_ca_wv   = (const float*)d_in[26];
  const float* dec_ca_wo   = (const float*)d_in[27];
  const float* dec_ff1_w   = (const float*)d_in[28];
  const float* dec_ff1_b   = (const float*)d_in[29];
  const float* dec_ff2_w   = (const float*)d_in[30];
  const float* dec_ff2_b   = (const float*)d_in[31];
  const float* dec_ln1_g   = (const float*)d_in[32];
  const float* dec_ln1_b   = (const float*)d_in[33];
  const float* dec_ln2_g   = (const float*)d_in[34];
  const float* dec_ln2_b   = (const float*)d_in[35];
  const float* dec_ln3_g   = (const float*)d_in[36];
  const float* dec_ln3_b   = (const float*)d_in[37];
  const float* proj_w      = (const float*)d_in[38];
  const float* proj_b      = (const float*)d_in[39];

  // workspace carve-up
  char* ws = (char*)d_ws;
  size_t off = 0;
  auto carve = [&](size_t bytes) -> void* {
    void* p = ws + off;
    off += (bytes + 255) & ~(size_t)255;
    return p;
  };
  float*          xe   = (float*)carve((size_t)M * D * 4);
  unsigned short* xeb  = (unsigned short*)carve((size_t)M * D * 2);
  float*          xd   = (float*)carve((size_t)M * D * 4);
  unsigned short* xdb  = (unsigned short*)carve((size_t)M * D * 2);
  unsigned short* qb   = (unsigned short*)carve((size_t)M * D * 2);
  unsigned short* kb   = (unsigned short*)carve((size_t)M * D * 2);
  unsigned short* vb   = (unsigned short*)carve((size_t)M * D * 2);
  unsigned short* vt   = (unsigned short*)carve((size_t)M * D * 2);
  float*          ctx  = (float*)carve((size_t)M * D * 4);
  unsigned short* ctxb = (unsigned short*)carve((size_t)M * D * 2);
  float*          af   = (float*)carve((size_t)M * D * 4);
  unsigned short* hb   = (unsigned short*)carve((size_t)M * F * 2);
  unsigned short* wt   = (unsigned short*)carve((size_t)F * D * 2);

  auto cvtW = [&](const float* W, int K, int N) {
    int tot = K * N;
    k_w_bf16t<<<(tot + 255) / 256, 256, 0, stream>>>(W, wt, K, N);
  };
  auto gemm = [&](const unsigned short* A, const float* bias,
                  const float* resid, float* of, unsigned short* ob,
                  int N, int K, int relu) {
    dim3 g(N / 64, M / 128);
    k_gemm_wmma<<<g, 256, 0, stream>>>(A, wt, bias, resid, of, ob, N, K, relu);
  };
  auto vtrans = [&]() {
    k_v_t<<<(M * D) / 256, 256, 0, stream>>>(vb, vt, L);
  };
  auto attn = [&](int kvlen) {
    k_attn<<<dim3(Bz * H, L / 128), 256, 0, stream>>>(qb, kb, vt, ctx, ctxb,
                                                      L, kvlen);
  };
  auto lnorm = [&](const float* g, const float* b, float* yf,
                   unsigned short* yb) {
    k_layernorm<<<M / 8, 256, 0, stream>>>(af, g, b, yf, yb);
  };

  // ---- input projections ----
  k_input_proj<<<(M * D) / 256, 256, 0, stream>>>(src, src_pos, src_lin_w,
                                                  src_lin_b, xe, xeb, 16);
  k_input_proj<<<(M * D) / 256, 256, 0, stream>>>(tgt, tgt_pos, tgt_lin_w,
                                                  tgt_lin_b, xd, xdb, 8);

  // ---- encoder ----
  for (int i = 0; i < 4; ++i) {
    const size_t wo = (size_t)i * D * D;
    cvtW(enc_wq + wo, D, D); gemm(xeb, nullptr, nullptr, nullptr, qb, D, D, 0);
    cvtW(enc_wk + wo, D, D); gemm(xeb, nullptr, nullptr, nullptr, kb, D, D, 0);
    cvtW(enc_wv + wo, D, D); gemm(xeb, nullptr, nullptr, nullptr, vb, D, D, 0);
    vtrans();
    attn(512);
    cvtW(enc_wo + wo, D, D); gemm(ctxb, nullptr, xe, af, nullptr, D, D, 0);
    lnorm(enc_ln1_g + i * D, enc_ln1_b + i * D, xe, xeb);
    cvtW(enc_ff1_w + (size_t)i * D * F, D, F);
    gemm(xeb, enc_ff1_b + (size_t)i * F, nullptr, nullptr, hb, F, D, 1);
    cvtW(enc_ff2_w + (size_t)i * F * D, F, D);
    gemm(hb, enc_ff2_b + (size_t)i * D, xe, af, nullptr, D, F, 0);
    lnorm(enc_ln2_g + i * D, enc_ln2_b + i * D, xe, xeb);
  }

  // ---- decoder ----
  for (int i = 0; i < 4; ++i) {
    const size_t wo = (size_t)i * D * D;
    // self-attention (column mask == effective kv length 416)
    cvtW(dec_sa_wq + wo, D, D); gemm(xdb, nullptr, nullptr, nullptr, qb, D, D, 0);
    cvtW(dec_sa_wk + wo, D, D); gemm(xdb, nullptr, nullptr, nullptr, kb, D, D, 0);
    cvtW(dec_sa_wv + wo, D, D); gemm(xdb, nullptr, nullptr, nullptr, vb, D, D, 0);
    vtrans();
    attn(512 - 96);
    cvtW(dec_sa_wo + wo, D, D); gemm(ctxb, nullptr, xd, af, nullptr, D, D, 0);
    lnorm(dec_ln1_g + i * D, dec_ln1_b + i * D, xd, xdb);
    // cross-attention (K/V from final encoder output)
    cvtW(dec_ca_wq + wo, D, D); gemm(xdb, nullptr, nullptr, nullptr, qb, D, D, 0);
    cvtW(dec_ca_wk + wo, D, D); gemm(xeb, nullptr, nullptr, nullptr, kb, D, D, 0);
    cvtW(dec_ca_wv + wo, D, D); gemm(xeb, nullptr, nullptr, nullptr, vb, D, D, 0);
    vtrans();
    attn(512);
    cvtW(dec_ca_wo + wo, D, D); gemm(ctxb, nullptr, xd, af, nullptr, D, D, 0);
    lnorm(dec_ln2_g + i * D, dec_ln2_b + i * D, xd, xdb);
    // FFN
    cvtW(dec_ff1_w + (size_t)i * D * F, D, F);
    gemm(xdb, dec_ff1_b + (size_t)i * F, nullptr, nullptr, hb, F, D, 1);
    cvtW(dec_ff2_w + (size_t)i * F * D, F, D);
    gemm(hb, dec_ff2_b + (size_t)i * D, xd, af, nullptr, D, F, 0);
    lnorm(dec_ln3_g + i * D, dec_ln3_b + i * D, xd, xdb);
  }

  // ---- final projection ----
  k_final_proj<<<(M * 16) / 256, 256, 0, stream>>>(xd, proj_w, proj_b,
                                                   (float*)d_out);
}
